// Block_46566035423950
// MI455X (gfx1250) — matrix operand order repaired and checked by reading, not compiled, not run on hardware
//
#include <hip/hip_runtime.h>
#include <hip/hip_bf16.h>

typedef __attribute__((ext_vector_type(16))) __bf16 v16bf;
typedef __attribute__((ext_vector_type(8)))  float  v8f;
typedef unsigned int   u32;
typedef unsigned short u16;

#define BN   30720   // B*N tokens
#define CDIM 1024
#define HS3  3072

static __device__ __forceinline__ u16 f2bf(float f) {
  u32 u = __float_as_uint(f);
  u32 r = u + 0x7FFFu + ((u >> 16) & 1u);   // round-to-nearest-even
  return (u16)(r >> 16);
}

// XOR lane shuffle via ds_swizzle (group-of-32 mode, immediate pattern):
// one instruction, no index VGPR, no bounds cndmask.
template<int XM>
static __device__ __forceinline__ float swz_xor(float v) {
  return __uint_as_float((u32)__builtin_amdgcn_ds_swizzle(
      (int)__float_as_uint(v), (XM << 10) | 0x1f));
}

// CDNA5 async global->LDS copy (GV mode): VDST = per-lane LDS byte address,
// VADDR = per-lane 64-bit global address. Tracked by ASYNCcnt.
static __device__ __forceinline__ void async_b128(void* lds, const void* gptr) {
  asm volatile("global_load_async_to_lds_b128 %0, %1, off"
               :: "v"((u32)(unsigned long long)(uintptr_t)lds),
                  "v"((unsigned long long)(uintptr_t)gptr)
               : "memory");
}

// ---------------- LayerNorm: f32 row (1024) -> bf16 ----------------
__global__ __launch_bounds__(256) void ln_bf16_kernel(
    const float* __restrict__ x, const float* __restrict__ g,
    const float* __restrict__ b, u16* __restrict__ y)
{
  __shared__ float red[16];
  const int row = blockIdx.x, t = threadIdx.x;
  const float4 v = ((const float4*)(x + (size_t)row * CDIM))[t];
  float s  = v.x + v.y + v.z + v.w;
  float ss = v.x*v.x + v.y*v.y + v.z*v.z + v.w*v.w;
  s += swz_xor<16>(s);  ss += swz_xor<16>(ss);
  s += swz_xor<8>(s);   ss += swz_xor<8>(ss);
  s += swz_xor<4>(s);   ss += swz_xor<4>(ss);
  s += swz_xor<2>(s);   ss += swz_xor<2>(ss);
  s += swz_xor<1>(s);   ss += swz_xor<1>(ss);
  if ((t & 31) == 0) { red[t >> 5] = s; red[8 + (t >> 5)] = ss; }
  __syncthreads();
  s = 0.f; ss = 0.f;
  #pragma unroll
  for (int i = 0; i < 8; ++i) { s += red[i]; ss += red[8 + i]; }
  const float mean = s * (1.f / CDIM);
  const float rstd = rsqrtf(ss * (1.f / CDIM) - mean * mean + 1e-5f);
  const float4 gv = ((const float4*)g)[t];
  const float4 bv = ((const float4*)b)[t];
  uint2 o;
  o.x = (u32)f2bf((v.x - mean) * rstd * gv.x + bv.x)
      | ((u32)f2bf((v.y - mean) * rstd * gv.y + bv.y) << 16);
  o.y = (u32)f2bf((v.z - mean) * rstd * gv.z + bv.z)
      | ((u32)f2bf((v.w - mean) * rstd * gv.w + bv.w) << 16);
  ((uint2*)(y + (size_t)row * CDIM))[t] = o;
}

// ---------------- Weight transpose: W[K][Nc] f32 -> Wt[Nc][K] bf16 ----------------
__global__ __launch_bounds__(256) void transpose_f32_to_bf16(
    const float* __restrict__ W, u16* __restrict__ Wt, int K, int Nc)
{
  __shared__ float tile[32][33];
  const int nt = blockIdx.x * 32, kt = blockIdx.y * 32;
  const int tx = threadIdx.x & 31, ty = threadIdx.x >> 5;
  #pragma unroll
  for (int i = 0; i < 4; ++i) {
    const int k = ty + i * 8;
    tile[k][tx] = W[(size_t)(kt + k) * Nc + nt + tx];
  }
  __syncthreads();
  #pragma unroll
  for (int i = 0; i < 4; ++i) {
    const int n = ty + i * 8;
    Wt[(size_t)(nt + n) * K + kt + tx] = f2bf(tile[tx][n]);
  }
}

// W_Iconv [64][15] f32 -> Wct [16][64] bf16 (transposed, row 15 zero-padded)
__global__ void build_wct(const float* __restrict__ Wc, u16* __restrict__ Wct) {
  const int i = blockIdx.x * 256 + threadIdx.x;
  if (i < 16 * 64) {
    const int n = i >> 6, k = i & 63;
    Wct[i] = f2bf(n < 15 ? Wc[k * 15 + n] : 0.f);
  }
}

// ---------------- bf16 WMMA GEMM: out[M][Nc] = A[M][K] @ Bt[Nc][K]^T ----------------
// Block tile 64x256 (8 waves, each 32x64 via 8 accumulators). K stepped by 32 with
// double-buffered async global->LDS staging.
// MODE 0: store bf16.  MODE 1: +bias +residual, store f32.  MODE 2: +bias, GELU, store bf16.
template<int MODE>
__global__ __launch_bounds__(256) void gemm_bf16_wmma(
    const u16* __restrict__ A, const u16* __restrict__ Bt,
    const float* __restrict__ bias, const float* __restrict__ res,
    void* __restrict__ outp, int K, int Nc)
{
  __shared__ __align__(16) u16 sA[2][64 * 32];    //  2 x 4 KB
  __shared__ __align__(16) u16 sB[2][256 * 32];   //  2 x 16 KB
  const int t = threadIdx.x, lane = t & 31, w = t >> 5;
  const int wm = w & 1, wn = w >> 1;
  const int m0 = blockIdx.x * 64, n0 = blockIdx.y * 256;

  const v8f vzero = {};
  v8f acc[2][4];
  #pragma unroll
  for (int mi = 0; mi < 2; ++mi)
    #pragma unroll
    for (int s = 0; s < 4; ++s) acc[mi][s] = vzero;

  // cooperative tile staging: A 64x32 (16B/thread), B 256x32 (row t, 4x16B/thread)
  const int arow = t >> 2, akc = (t & 3) * 8;
  const u16* gA = A + (size_t)(m0 + arow) * K + akc;
  const u16* gB = Bt + (size_t)(n0 + t) * K;
  u16* lA[2] = { &sA[0][arow * 32 + akc], &sA[1][arow * 32 + akc] };
  u16* lB[2] = { &sB[0][t * 32],          &sB[1][t * 32] };

  const int kb4 = (lane >> 4) * 4;   // A-fragment dword base within 16-dword row
  const int kb8 = (lane >> 4) * 4;   // B-fragment dword base
  const int rl = lane & 15;

  // issue 5 async b128 transfers per thread into buffer `buf` for K-step k0
  auto issue = [&](int buf, int k0) {
    async_b128(lA[buf], gA + k0);
    #pragma unroll
    for (int j = 0; j < 4; ++j) async_b128(lB[buf] + j * 8, gB + k0 + j * 8);
  };

  const int nk = K >> 5;
  issue(0, 0);
  for (int i = 0; i < nk; ++i) {
    const int cur = i & 1;
    if (i + 1 < nk) {
      issue(cur ^ 1, (i + 1) * 32);
      // retire the 5 older transfers (this buffer); keep the 5 new in flight
      asm volatile("s_wait_asynccnt 0x5" ::: "memory");
    } else {
      asm volatile("s_wait_asynccnt 0x0" ::: "memory");
    }
    __syncthreads();

    const u32* sAu = (const u32*)sA[cur];
    const u32* sBu = (const u32*)sB[cur];
    union { v16bf v; u32 u[8]; } fa[2];
    #pragma unroll
    for (int mi = 0; mi < 2; ++mi) {
      const int ra = (wm * 32 + mi * 16 + rl) * 16;
      *(uint4*)&fa[mi].u[0] = *(const uint4*)(sAu + ra + kb4);
      *(uint4*)&fa[mi].u[4] = *(const uint4*)(sAu + ra + 8 + kb4);
    }
    #pragma unroll
    for (int s = 0; s < 4; ++s) {
      union { v16bf v; u32 u[8]; } fb;
      const int rb = (wn * 64 + s * 16 + rl) * 16;
      *(uint4*)&fb.u[0] = *(const uint4*)(sBu + rb + kb8);
      *(uint4*)&fb.u[4] = *(const uint4*)(sBu + rb + kb8 + 8);
      acc[0][s] = __builtin_amdgcn_wmma_f32_16x16x32_bf16(
          false, fa[0].v, false, fb.v, (short)0, acc[0][s], false, false);
      acc[1][s] = __builtin_amdgcn_wmma_f32_16x16x32_bf16(
          false, fa[1].v, false, fb.v, (short)0, acc[1][s], false, false);
    }
    __syncthreads();   // all waves done reading `cur` before it is refilled
  }

  const int mh = (lane >> 4) * 8;
  #pragma unroll
  for (int mi = 0; mi < 2; ++mi) {
    #pragma unroll
    for (int s = 0; s < 4; ++s) {
      const int n = n0 + wn * 64 + s * 16 + rl;
      const float bv = (MODE != 0) ? bias[n] : 0.f;
      #pragma unroll
      for (int r = 0; r < 8; ++r) {
        const int m = m0 + wm * 32 + mi * 16 + mh + r;
        float v = acc[mi][s][r];
        if (MODE == 1) {
          v += bv + res[(size_t)m * Nc + n];
          ((float*)outp)[(size_t)m * Nc + n] = v;
        } else if (MODE == 2) {
          v += bv;
          v = 0.5f * v * (1.f + erff(v * 0.70710678118f));   // exact GELU
          ((u16*)outp)[(size_t)m * Nc + n] = f2bf(v);
        } else {
          ((u16*)outp)[(size_t)m * Nc + n] = f2bf(v);
        }
      }
    }
  }
}

// ---------------- Attention per (b,h): one wave, WMMA scores + softmax + P@Jv ----------------
// Padded row (m=15) uses a clamped address; its results are never stored. Padded score
// column (n=15) is masked with -inf before softmax; padded K columns of P are zeroed in LDS.
__global__ __launch_bounds__(32) void attn_kernel(
    const u16* __restrict__ Jqkv, const u16* __restrict__ Iqkv,
    const u16* __restrict__ Wct, u16* __restrict__ ctx)
{
  __shared__ __align__(16) u16 sP[16 * 32];    // P padded to 16x32 (K for P@Jv)
  __shared__ __align__(16) u16 sVT[64 * 32];   // Jv^T [d][token], token padded to 32
  __shared__ __align__(16) u16 xbuf[16 * 64];  // output x tile staging
  const int l = threadIdx.x;
  const int b = blockIdx.x >> 4, h = blockIdx.x & 15;
  u32* sPu = (u32*)sP;
  u32* sVu = (u32*)sVT;
  for (int i = l; i < 256;  i += 32) sPu[i] = 0u;
  for (int i = l; i < 1024; i += 32) sVu[i] = 0u;

  const size_t tok0 = (size_t)b * 15;
  // Jv^T into LDS: 120 b128 chunks (15 tokens x 8 chunks of 8 elems)
  for (int c = l; c < 120; c += 32) {
    const int tk = c >> 3, dc = (c & 7) * 8;
    union { uint4 q; u16 e[8]; } u;
    u.q = *(const uint4*)(Jqkv + (tok0 + tk) * HS3 + 2048 + h * 64 + dc);
    #pragma unroll
    for (int j = 0; j < 8; ++j) sVT[(dc + j) * 32 + tk] = u.e[j];
  }

  const int nrow = l & 15;
  const int rowc = nrow < 15 ? nrow : 0;     // clamp padded row/col address
  const int kb4 = (l >> 4) * 4, kb8 = (l >> 4) * 4;
  const u32* qJ = (const u32*)(Jqkv + (tok0 + rowc) * HS3 + h * 64);
  const u32* kJ = (const u32*)(Jqkv + (tok0 + rowc) * HS3 + 1024 + h * 64);
  const u32* qI = (const u32*)(Iqkv + (tok0 + rowc) * HS3 + h * 64);
  const u32* kI = (const u32*)(Iqkv + (tok0 + rowc) * HS3 + 1024 + h * 64);
  const u32* vI = (const u32*)(Iqkv + (tok0 + rowc) * HS3 + 2048 + h * 64);
  const u32* wc = (const u32*)(Wct + nrow * 64);

  v8f acc = {};
  #pragma unroll
  for (int kh = 0; kh < 2; ++kh) {           // HS=64 -> two K=32 steps
    const int kA = kh * 16 + kb4;
    const int kB = kh * 16 + kb8;
    union { v16bf v; u32 u[8]; } a, bb;
    // Jq @ Jk^T
    *(uint4*)&a.u[0]  = *(const uint4*)(qJ + kA);
    *(uint4*)&a.u[4]  = *(const uint4*)(qJ + kA + 8);
    *(uint4*)&bb.u[0] = *(const uint4*)(kJ + kB);
    *(uint4*)&bb.u[4] = *(const uint4*)(kJ + kB + 8);
    acc = __builtin_amdgcn_wmma_f32_16x16x32_bf16(false, a.v, false, bb.v, (short)0, acc, false, false);
    // Iq @ Ik^T
    *(uint4*)&a.u[0]  = *(const uint4*)(qI + kA);
    *(uint4*)&a.u[4]  = *(const uint4*)(qI + kA + 8);
    *(uint4*)&bb.u[0] = *(const uint4*)(kI + kB);
    *(uint4*)&bb.u[4] = *(const uint4*)(kI + kB + 8);
    acc = __builtin_amdgcn_wmma_f32_16x16x32_bf16(false, a.v, false, bb.v, (short)0, acc, false, false);
    // Iv @ W_Iconv  (Wct row 15 is zero-padded)
    *(uint4*)&a.u[0]  = *(const uint4*)(vI + kA);
    *(uint4*)&a.u[4]  = *(const uint4*)(vI + kA + 8);
    *(uint4*)&bb.u[0] = *(const uint4*)(wc + kB);
    *(uint4*)&bb.u[4] = *(const uint4*)(wc + kB + 8);
    acc = __builtin_amdgcn_wmma_f32_16x16x32_bf16(false, a.v, false, bb.v, (short)0, acc, false, false);
  }

  // Softmax per row: each C row lives in one 16-lane group (lane = column n).
  // XOR masks < 16 keep the reduction confined to each 16-lane group.
  float p[8];
  #pragma unroll
  for (int r = 0; r < 8; ++r) {
    float s = acc[r] * 0.125f;               // SCALE = 64^-0.5
    if (nrow == 15) s = -1e30f;              // mask padded column
    float mx = s;
    mx = fmaxf(mx, swz_xor<8>(mx));
    mx = fmaxf(mx, swz_xor<4>(mx));
    mx = fmaxf(mx, swz_xor<2>(mx));
    mx = fmaxf(mx, swz_xor<1>(mx));
    const float e = __expf(s - mx);
    float sm = e;
    sm += swz_xor<8>(sm);
    sm += swz_xor<4>(sm);
    sm += swz_xor<2>(sm);
    sm += swz_xor<1>(sm);
    p[r] = e / sm;
  }

  __syncthreads();
  const int mh = (l >> 4) * 8;
  #pragma unroll
  for (int r = 0; r < 8; ++r) sP[(mh + r) * 32 + nrow] = f2bf(p[r]);
  __syncthreads();

  // x = P @ Jv  (A = P from LDS, B = Jv^T rows from LDS); stage x in LDS for b128 stores
  union { v16bf v; u32 u[8]; } a2;
  *(uint4*)&a2.u[0] = *(const uint4*)(sPu + nrow * 16 + kb4);
  *(uint4*)&a2.u[4] = *(const uint4*)(sPu + nrow * 16 + 8 + kb4);
  #pragma unroll
  for (int dt = 0; dt < 4; ++dt) {
    union { v16bf v; u32 u[8]; } b2;
    const int d = dt * 16 + nrow;
    *(uint4*)&b2.u[0] = *(const uint4*)(sVu + d * 16 + kb8);
    *(uint4*)&b2.u[4] = *(const uint4*)(sVu + d * 16 + kb8 + 8);
    v8f x = {};
    x = __builtin_amdgcn_wmma_f32_16x16x32_bf16(false, a2.v, false, b2.v, (short)0, x, false, false);
    #pragma unroll
    for (int r = 0; r < 8; ++r)
      xbuf[(mh + r) * 64 + dt * 16 + nrow] = f2bf(x[r]);   // row 15 staged, never stored
  }
  __syncthreads();
  // coalesced stores: 15 rows x 64 bf16 = 120 b128 chunks
  for (int c = l; c < 120; c += 32) {
    const int tk = c >> 3, dc = (c & 7) * 8;
    *(uint4*)(ctx + (tok0 + tk) * CDIM + h * 64 + dc) =
        *(const uint4*)(xbuf + tk * 64 + dc);
  }
}

extern "C" void kernel_launch(void* const* d_in, const int* in_sizes, int n_in,
                              void* d_out, int out_size, void* d_ws, size_t ws_size,
                              hipStream_t stream)
{
  const float* joint_in = (const float*)d_in[0];
  const float* rel_in   = (const float*)d_in[1];
  const float* W_Jqkv   = (const float*)d_in[2];
  const float* W_Iqk    = (const float*)d_in[3];
  const float* W_Iconv  = (const float*)d_in[4];
  const float* W_proj   = (const float*)d_in[5];
  const float* b_proj   = (const float*)d_in[6];
  const float* g_attn1  = (const float*)d_in[7];
  const float* b_attn1  = (const float*)d_in[8];
  const float* g_attn2  = (const float*)d_in[9];
  const float* b_attn2  = (const float*)d_in[10];
  const float* g_joint  = (const float*)d_in[11];
  const float* b_joint  = (const float*)d_in[12];
  const float* W_fc1    = (const float*)d_in[13];
  const float* b_fc1    = (const float*)d_in[14];
  const float* W_fc2    = (const float*)d_in[15];
  const float* b_fc2    = (const float*)d_in[16];

  char* ws = (char*)d_ws;
  size_t off = 0;
  auto alloc = [&](size_t bytes) -> void* {
    off = (off + 255) & ~(size_t)255;
    void* p = ws + off;
    off += bytes;
    return p;
  };
  u16*   xj    = (u16*)alloc((size_t)BN * CDIM * 2);   // LN(joint); reused as yln
  u16*   xi    = (u16*)alloc((size_t)BN * CDIM * 2);   // LN(relation); reused as h1
  u16*   WJt   = (u16*)alloc((size_t)HS3 * CDIM * 2);
  u16*   WIt   = (u16*)alloc((size_t)HS3 * CDIM * 2);
  u16*   Wpt   = (u16*)alloc((size_t)CDIM * CDIM * 2);
  u16*   W1t   = (u16*)alloc((size_t)512 * CDIM * 2);
  u16*   W2t   = (u16*)alloc((size_t)CDIM * 512 * 2);
  u16*   Wct   = (u16*)alloc((size_t)16 * 64 * 2);
  u16*   Jqkv  = (u16*)alloc((size_t)BN * HS3 * 2);
  u16*   Iqkv  = (u16*)alloc((size_t)BN * HS3 * 2);
  u16*   ctx   = (u16*)alloc((size_t)BN * CDIM * 2);
  float* joint = (float*)alloc((size_t)BN * CDIM * 4);

  // Weight prep (transpose to [out,in] bf16 so B-fragments are K-contiguous)
  transpose_f32_to_bf16<<<dim3(3072/32, 1024/32), 256, 0, stream>>>(W_Jqkv, WJt, 1024, 3072);
  transpose_f32_to_bf16<<<dim3(3072/32, 1024/32), 256, 0, stream>>>(W_Iqk,  WIt, 1024, 3072);
  transpose_f32_to_bf16<<<dim3(1024/32, 1024/32), 256, 0, stream>>>(W_proj, Wpt, 1024, 1024);
  transpose_f32_to_bf16<<<dim3( 512/32, 1024/32), 256, 0, stream>>>(W_fc1,  W1t, 1024,  512);
  transpose_f32_to_bf16<<<dim3(1024/32,  512/32), 256, 0, stream>>>(W_fc2,  W2t,  512, 1024);
  build_wct<<<4, 256, 0, stream>>>(W_Iconv, Wct);

  // Pre-attention LayerNorms
  ln_bf16_kernel<<<BN, 256, 0, stream>>>(joint_in, g_attn1, b_attn1, xj);
  ln_bf16_kernel<<<BN, 256, 0, stream>>>(rel_in,   g_attn2, b_attn2, xi);

  // QKV projections (compute-bound: bf16 WMMA, async double-buffered staging)
  gemm_bf16_wmma<0><<<dim3(BN/64, 3072/256), 256, 0, stream>>>(xj, WJt, nullptr, nullptr, Jqkv, 1024, 3072);
  gemm_bf16_wmma<0><<<dim3(BN/64, 3072/256), 256, 0, stream>>>(xi, WIt, nullptr, nullptr, Iqkv, 1024, 3072);

  // Fused attention (scores + softmax + context) per (b,h)
  attn_kernel<<<2048 * 16, 32, 0, stream>>>(Jqkv, Iqkv, Wct, ctx);

  // Output projection + bias + residual -> joint (f32)
  gemm_bf16_wmma<1><<<dim3(BN/64, 1024/256), 256, 0, stream>>>(ctx, Wpt, b_proj, joint_in, joint, 1024, 1024);

  // MLP
  u16* yln = xj;
  ln_bf16_kernel<<<BN, 256, 0, stream>>>(joint, g_joint, b_joint, yln);
  u16* h1 = xi;
  gemm_bf16_wmma<2><<<dim3(BN/64,  512/256), 256, 0, stream>>>(yln, W1t, b_fc1, nullptr, h1, 1024, 512);
  gemm_bf16_wmma<1><<<dim3(BN/64, 1024/256), 256, 0, stream>>>(h1,  W2t, b_fc2, joint, (float*)d_out, 512, 1024);

  (void)in_sizes; (void)n_in; (void)out_size; (void)ws_size;
}
